// MemLayer_65386582114560
// MI455X (gfx1250) — matrix operand order repaired and checked by reading, not compiled, not run on hardware
//
#include <hip/hip_runtime.h>
#include <stdint.h>

// ---------------------------------------------------------------------------
// MemLayer: out[b,o] = -(mean_d (x[b,d]-w[o,d])^2) + bias[o]
//         = (2/D)*(X W^T)[b,o] - |x_b|^2/D - (|w_o|^2/D - bias[o])
// GEMM done with bf16x3 split-precision on V_WMMA_F32_16X16X32_BF16 (wave32),
// software-pipelined global loads; epilogue vectors staged to LDS with
// GLOBAL_LOAD_ASYNC_TO_LDS (ASYNCcnt), waited only before the epilogue.
// ---------------------------------------------------------------------------

typedef __bf16 v16bf __attribute__((ext_vector_type(16)));
typedef float  v8f   __attribute__((ext_vector_type(8)));
typedef int    i32x4 __attribute__((ext_vector_type(4)));

#define D_IN 1024
#define B_SZ 8192
#define O_SZ 4096

#define BM 128
#define BN 128
#define BK 32
#define NTHREADS 256

// ----- bf16 helpers (integer RNE, no scalar __bf16 arithmetic) --------------
static __device__ __forceinline__ uint32_t f32_to_bf16_rne(float f) {
  uint32_t u = __builtin_bit_cast(uint32_t, f);
  u += 0x7FFFu + ((u >> 16) & 1u);
  return u >> 16; // low 16 bits hold the bf16
}
static __device__ __forceinline__ float bf16_to_f32(uint32_t h16) {
  return __builtin_bit_cast(float, h16 << 16);
}
static __device__ __forceinline__ uint32_t pack2(uint32_t k0, uint32_t k1) {
  // even K in [15:0], odd K in [31:16] (matches WMMA packed-pair layout)
  return (k1 << 16) | (k0 & 0xFFFFu);
}

// ----- async global -> LDS copy (16B per lane), tracked by ASYNCcnt ---------
static __device__ __forceinline__ void async_copy_b128(void* lds_dst,
                                                       const void* gbl_src) {
#if __has_builtin(__builtin_amdgcn_global_load_async_to_lds_b128)
  __builtin_amdgcn_global_load_async_to_lds_b128(
      (__attribute__((address_space(1))) i32x4*)gbl_src,
      (__attribute__((address_space(3))) i32x4*)lds_dst,
      0, 0);
#else
  asm volatile("global_load_async_to_lds_b128 %0, %1, off"
               :
               : "v"((uint32_t)(uintptr_t)lds_dst),
                 "v"((uint64_t)(uintptr_t)gbl_src)
               : "memory");
#endif
}
static __device__ __forceinline__ void wait_asynccnt0() {
#if __has_builtin(__builtin_amdgcn_s_wait_asynccnt)
  __builtin_amdgcn_s_wait_asynccnt(0);
#else
  asm volatile("s_wait_asynccnt 0x0" ::: "memory");
#endif
}

// ---------------------------------------------------------------------------
// r[b] = sum_d x[b,d]^2 / D        (one wave per row, 8 rows per block)
// ---------------------------------------------------------------------------
__global__ __launch_bounds__(256) void row_sumsq_kernel(
    const float* __restrict__ X, float* __restrict__ r) {
  const int wid  = threadIdx.x >> 5;
  const int lane = threadIdx.x & 31;
  const int row  = blockIdx.x * 8 + wid;
  const float4* p = (const float4*)(X + (size_t)row * D_IN);
  float s = 0.f;
#pragma unroll
  for (int i = 0; i < (D_IN / 4) / 32; ++i) {
    float4 v = p[lane + i * 32];
    s += v.x * v.x + v.y * v.y + v.z * v.z + v.w * v.w;
  }
#pragma unroll
  for (int m = 16; m > 0; m >>= 1) s += __shfl_xor(s, m, 32);
  if (lane == 0) r[row] = s * (1.0f / (float)D_IN);
}

// ---------------------------------------------------------------------------
// c[o] = sum_d w[o,d]^2 / D - bias[o]
// ---------------------------------------------------------------------------
__global__ __launch_bounds__(256) void col_term_kernel(
    const float* __restrict__ W, const float* __restrict__ bias,
    float* __restrict__ c) {
  const int wid  = threadIdx.x >> 5;
  const int lane = threadIdx.x & 31;
  const int row  = blockIdx.x * 8 + wid;
  const float4* p = (const float4*)(W + (size_t)row * D_IN);
  float s = 0.f;
#pragma unroll
  for (int i = 0; i < (D_IN / 4) / 32; ++i) {
    float4 v = p[lane + i * 32];
    s += v.x * v.x + v.y * v.y + v.z * v.z + v.w * v.w;
  }
#pragma unroll
  for (int m = 16; m > 0; m >>= 1) s += __shfl_xor(s, m, 32);
  if (lane == 0) c[row] = s * (1.0f / (float)D_IN) - bias[row];
}

// ---------------------------------------------------------------------------
// Main GEMM: 128x128 block tile, BK=32, 8 waves; wave = 64x32 (4x2 wmma tiles)
// LDS holds A/B fragments pre-converted & pre-swizzled into WMMA VGPR layout:
//   A 16x32 bf16: lane(l<16)->M=l, vgpr v pair K: v0..3 -> K{0..7},
//                 v4..7 -> K{16..23}; lanes 16..31 get K+8.
//   B 32x16 bf16: same K placement as A (v0..3 -> K{0..7}, v4..7 -> K{16..23}, lanes 16..31 K+8),
//                 N = lane&15.
// Pipeline per chunk: [convert+LDS store] sync [issue next global loads +
//                     prefetch k+2] [ds fragment loads + 24 WMMA] sync
// ---------------------------------------------------------------------------
__global__ __launch_bounds__(NTHREADS) void memlayer_gemm_kernel(
    const float* __restrict__ X, const float* __restrict__ W,
    const float* __restrict__ rB, const float* __restrict__ cO,
    float* __restrict__ out) {
  __shared__ __attribute__((aligned(32))) uint32_t aHi[8][32][8];
  __shared__ __attribute__((aligned(32))) uint32_t aLo[8][32][8];
  __shared__ __attribute__((aligned(32))) uint32_t bHi[8][32][8];
  __shared__ __attribute__((aligned(32))) uint32_t bLo[8][32][8];
  __shared__ __attribute__((aligned(16))) float rS[BM];
  __shared__ __attribute__((aligned(16))) float cS[BN];

  const int tid   = threadIdx.x;
  const int bRow0 = blockIdx.y * BM;
  const int oCol0 = blockIdx.x * BN;

  // ---- epilogue vectors: async global->LDS, rides under the whole K loop --
  if (tid < 32) {
    async_copy_b128(&rS[tid * 4], &rB[bRow0 + tid * 4]);       // 32 x 16B
  } else if (tid < 64) {
    const int t = tid - 32;
    async_copy_b128(&cS[t * 4], &cO[oCol0 + t * 4]);           // 32 x 16B
  }

  const int lane = tid & 31;
  const int wid  = tid >> 5;
  const int wm   = wid >> 2; // 0..1  (M direction, 64 rows each)
  const int wn   = wid & 3;  // 0..3  (N direction, 32 cols each)

  v8f acc[4][2];
#pragma unroll
  for (int mi = 0; mi < 4; ++mi)
#pragma unroll
    for (int ni = 0; ni < 2; ++ni)
      acc[mi][ni] = (v8f){0.f, 0.f, 0.f, 0.f, 0.f, 0.f, 0.f, 0.f};

  // loader decomposition: 256 threads cover 128 rows x 16 K-pairs, 8 row-steps
  const int lkp = tid & 15;  // K-pair index 0..15  (K = 2*lkp .. 2*lkp+1)
  const int lr0 = tid >> 4;  // row within 16-row group (== M within tile)

  // A-fragment placement for pair lkp
  const int aLane = lr0 + (((lkp >> 2) & 1) << 4);
  const int aV    = ((lkp >> 3) << 2) | (lkp & 3);
  // B-fragment placement for pair lkp
  const int bLane = lr0 + (((lkp >> 2) & 1) << 4);
  const int bV    = ((lkp >> 3) << 2) | (lkp & 3);

  // per-thread global base pointers (row = lr0 + 16*i, K offset = 2*lkp)
  const float* xPtr = X + (size_t)(bRow0 + lr0) * D_IN + 2 * lkp;
  const float* wPtr = W + (size_t)(oCol0 + lr0) * D_IN + 2 * lkp;

  // ---- preload chunk 0 into registers -------------------------------------
  float2 xr[8], wr[8];
#pragma unroll
  for (int i = 0; i < 8; ++i) {
    xr[i] = *(const float2*)(xPtr + (size_t)i * 16 * D_IN);
    wr[i] = *(const float2*)(wPtr + (size_t)i * 16 * D_IN);
  }

  for (int k0 = 0; k0 < D_IN; k0 += BK) {
    // ---- convert staged registers, store fragments to LDS -----------------
#pragma unroll
    for (int i = 0; i < 8; ++i) {
      uint32_t h0 = f32_to_bf16_rne(xr[i].x);
      uint32_t h1 = f32_to_bf16_rne(xr[i].y);
      uint32_t l0 = f32_to_bf16_rne(xr[i].x - bf16_to_f32(h0));
      uint32_t l1 = f32_to_bf16_rne(xr[i].y - bf16_to_f32(h1));
      aHi[i][aLane][aV] = pack2(h0, h1);
      aLo[i][aLane][aV] = pack2(l0, l1);
      uint32_t g0 = f32_to_bf16_rne(wr[i].x);
      uint32_t g1 = f32_to_bf16_rne(wr[i].y);
      uint32_t m0 = f32_to_bf16_rne(wr[i].x - bf16_to_f32(g0));
      uint32_t m1 = f32_to_bf16_rne(wr[i].y - bf16_to_f32(g1));
      bHi[i][bLane][bV] = pack2(g0, g1);
      bLo[i][bLane][bV] = pack2(m0, m1);
    }
    __syncthreads(); // fragments visible to all waves

    // ---- issue next chunk's global loads; waited on next iteration --------
    if (k0 + BK < D_IN) {
      const float* xn  = xPtr + k0 + BK;
      const float* wn2 = wPtr + k0 + BK;
#pragma unroll
      for (int i = 0; i < 8; ++i) {
        xr[i] = *(const float2*)(xn + (size_t)i * 16 * D_IN);
        wr[i] = *(const float2*)(wn2 + (size_t)i * 16 * D_IN);
      }
      // warm L2 for chunk k+2 (global_prefetch_b8)
      if (k0 + 2 * BK < D_IN) {
        __builtin_prefetch(xn + BK, 0, 2);
        __builtin_prefetch(wn2 + BK + (size_t)4 * 16 * D_IN, 0, 2);
      }
    }

    // ---- fragment loads + WMMA block --------------------------------------
    v16bf ah[4], al[4], bh[2], bl[2];
#pragma unroll
    for (int mi = 0; mi < 4; ++mi) {
      const int mt = wm * 4 + mi;
      ah[mi] = *(const v16bf*)&aHi[mt][lane][0];
      al[mi] = *(const v16bf*)&aLo[mt][lane][0];
    }
#pragma unroll
    for (int ni = 0; ni < 2; ++ni) {
      const int nt = wn * 2 + ni;
      bh[ni] = *(const v16bf*)&bHi[nt][lane][0];
      bl[ni] = *(const v16bf*)&bLo[nt][lane][0];
    }

#pragma unroll
    for (int mi = 0; mi < 4; ++mi) {
#pragma unroll
      for (int ni = 0; ni < 2; ++ni) {
        // bf16x3: hi*hi + hi*lo + lo*hi  (error ~2^-16, f32 accumulate)
        acc[mi][ni] = __builtin_amdgcn_wmma_f32_16x16x32_bf16(
            false, ah[mi], false, bh[ni], (short)0, acc[mi][ni], false, false);
        acc[mi][ni] = __builtin_amdgcn_wmma_f32_16x16x32_bf16(
            false, ah[mi], false, bl[ni], (short)0, acc[mi][ni], false, false);
        acc[mi][ni] = __builtin_amdgcn_wmma_f32_16x16x32_bf16(
            false, al[mi], false, bh[ni], (short)0, acc[mi][ni], false, false);
      }
    }
    __syncthreads(); // compute done before LDS overwrite next iteration
  }

  // ---- drain async rS/cS copies; make them visible to all waves -----------
  wait_asynccnt0();
  __syncthreads();

  // ---- epilogue: out = (2/D)*acc - rS[row] - cS[col], NT stores -----------
  const float scale = 2.0f / (float)D_IN;
  const int mHalf = (lane >> 4) * 8; // C/D layout: lanes 16..31 -> M+8
  const int nIdx  = lane & 15;
#pragma unroll
  for (int mi = 0; mi < 4; ++mi) {
#pragma unroll
    for (int ni = 0; ni < 2; ++ni) {
      const int colL = wn * 32 + ni * 16 + nIdx;
      const float cadd = cS[colL];
      const size_t col = (size_t)(oCol0 + colL);
#pragma unroll
      for (int r = 0; r < 8; ++r) {
        const int rowL = wm * 64 + mi * 16 + mHalf + r;
        const float val = scale * acc[mi][ni][r] - rS[rowL] - cadd;
        __builtin_nontemporal_store(val,
            &out[(size_t)(bRow0 + rowL) * O_SZ + col]);
      }
    }
  }
}

// ---------------------------------------------------------------------------
extern "C" void kernel_launch(void* const* d_in, const int* in_sizes, int n_in,
                              void* d_out, int out_size, void* d_ws,
                              size_t ws_size, hipStream_t stream) {
  const float* x    = (const float*)d_in[0]; // [8192,1024]
  const float* w    = (const float*)d_in[1]; // [4096,1024]
  const float* bias = (const float*)d_in[2]; // [4096]
  float* out = (float*)d_out;                // [8192,4096]

  float* rB = (float*)d_ws;       // |x_b|^2 / D          (8192 floats)
  float* cO = rB + B_SZ;          // |w_o|^2 / D - bias   (4096 floats)

  row_sumsq_kernel<<<B_SZ / 8, 256, 0, stream>>>(x, rB);
  col_term_kernel<<<O_SZ / 8, 256, 0, stream>>>(w, bias, cO);

  dim3 grid(O_SZ / BN, B_SZ / BM); // 32 x 64
  memlayer_gemm_kernel<<<grid, NTHREADS, 0, stream>>>(x, w, rB, cO, out);
}
